// GNNLayer_48756468744911
// MI455X (gfx1250) — matrix operand order repaired and checked by reading, not compiled, not run on hardware
//
#include <hip/hip_runtime.h>

typedef __bf16 bf16;
typedef __attribute__((ext_vector_type(16))) __bf16 v16bf;
typedef __attribute__((ext_vector_type(8)))  __bf16 v8bf;
typedef __attribute__((ext_vector_type(4)))  __bf16 v4bf;
typedef __attribute__((ext_vector_type(8)))  float   v8f;
typedef __attribute__((ext_vector_type(4)))  float   v4f;

constexpr int DIM   = 128;   // D
constexpr int OUT   = 128;   // O
constexpr int K3    = 384;   // 3*D
constexpr int KS_M  = 12;    // 384/32 k-steps, message GEMM
constexpr int KS_U  = 4;     // 128/32 k-steps, update GEMM
constexpr int TILE_E = 64;   // edges per workgroup
constexpr int LDA_M = K3 + 8;   // padded LDS stride (bf16): 196 dwords -> conflict-free
constexpr int LDA_U = OUT + 8;  // 68 dwords -> conflict-free
constexpr int PM_ELEMS = 8 * KS_M * 32 * 16;  // packed W_msg bf16 elems
constexpr int PU_ELEMS = 8 * KS_U * 32 * 16;  // packed W_upd bf16 elems

__global__ void zero_kernel(float* p, long long n) {
    long long i = (long long)blockIdx.x * blockDim.x + threadIdx.x;
    if (i < n) p[i] = 0.0f;
}

// Pack f32 weights into bf16 WMMA B-fragment order:
// p = ((cb*KS + ks)*32 + lane)*16 + i  holds  W[(ks*32 + (i>>3)*16 + (lane>>4)*8 + (i&7)) * OUT + cb*16 + (lane&15)]
__global__ void pack_weights(const float* __restrict__ Wmsg, const float* __restrict__ Wupd,
                             bf16* __restrict__ Pmsg, bf16* __restrict__ Pupd) {
    int t = blockIdx.x * 256 + threadIdx.x;
    if (t < PM_ELEMS) {
        int p = t;
        int cb = p / (KS_M * 512);
        int r  = p % (KS_M * 512);
        int ks = r / 512;
        int r2 = r % 512;
        int lane = r2 >> 4, i = r2 & 15;
        int col = cb * 16 + (lane & 15);
        int k   = ks * 32 + (i >> 3) * 16 + (lane >> 4) * 8 + (i & 7);
        Pmsg[p] = (bf16)Wmsg[k * OUT + col];
    } else if (t < PM_ELEMS + PU_ELEMS) {
        int p = t - PM_ELEMS;
        int cb = p / (KS_U * 512);
        int r  = p % (KS_U * 512);
        int ks = r / 512;
        int r2 = r % 512;
        int lane = r2 >> 4, i = r2 & 15;
        int col = cb * 16 + (lane & 15);
        int k   = ks * 32 + (i >> 3) * 16 + (lane >> 4) * 8 + (i & 7);
        Pupd[p] = (bf16)Wupd[k * OUT + col];
    }
}

__global__ void count_kernel(const int* __restrict__ dst, float* __restrict__ counts, int E_) {
    int e = blockIdx.x * 256 + threadIdx.x;
    if (e < E_) atomicAdd(&counts[dst[e]], 1.0f);
}

// Per workgroup: 64 edges x 128 outputs, batch = blockIdx.y. 8 waves, 1 colblock each.
__global__ void __launch_bounds__(256) msg_kernel(
        const float* __restrict__ x, const int* __restrict__ src, const int* __restrict__ dst,
        const float* __restrict__ eattr, const bf16* __restrict__ Pmsg,
        const float* __restrict__ bmsg, float* __restrict__ agg, int N_, int E_) {
    __shared__ int  s_src[TILE_E];
    __shared__ int  s_dst[TILE_E];
    __shared__ bf16 sA[TILE_E * LDA_M];

    const int tid = threadIdx.x;
    const int b   = blockIdx.y;
    const long long ebase = (long long)blockIdx.x * TILE_E;
    const bool full = (ebase + TILE_E) <= (long long)E_;   // uniform: all but last tile

    if (tid < TILE_E) {
        long long e = ebase + tid;
        int sv = 0, dv = 0;
        if (e < E_) { sv = src[e]; dv = dst[e]; }
        s_src[tid] = sv; s_dst[tid] = dv;
    }
    __syncthreads();

    // Stage A tile: 64 rows x 384 f32 -> bf16 LDS (6144 float4 chunks / 256 threads)
    const long long xbase = (long long)b * N_ * DIM;
    if (full) {
#pragma unroll
        for (int it = 0; it < 24; ++it) {
            int q   = it * 256 + tid;
            int row = q / 96;           // 96 float4 per row
            int c4  = q - row * 96;
            int col = c4 * 4;
            long long e = ebase + row;
            v4f v;
            if (col < 128)      v = *(const v4f*)(x + xbase + (long long)s_src[row] * DIM + col);
            else if (col < 256) v = *(const v4f*)(x + xbase + (long long)s_dst[row] * DIM + (col - 128));
            else                v = __builtin_nontemporal_load((const v4f*)(eattr + e * DIM + (col - 256)));
            v4bf o; o[0] = (bf16)v[0]; o[1] = (bf16)v[1]; o[2] = (bf16)v[2]; o[3] = (bf16)v[3];
            *(v4bf*)&sA[row * LDA_M + col] = o;
        }
    } else {
#pragma unroll
        for (int it = 0; it < 24; ++it) {
            int q   = it * 256 + tid;
            int row = q / 96;
            int c4  = q - row * 96;
            int col = c4 * 4;
            long long e = ebase + row;
            v4f v = (v4f)(0.0f);
            if (e < E_) {
                if (col < 128)      v = *(const v4f*)(x + xbase + (long long)s_src[row] * DIM + col);
                else if (col < 256) v = *(const v4f*)(x + xbase + (long long)s_dst[row] * DIM + (col - 128));
                else                v = __builtin_nontemporal_load((const v4f*)(eattr + e * DIM + (col - 256)));
            }
            v4bf o; o[0] = (bf16)v[0]; o[1] = (bf16)v[1]; o[2] = (bf16)v[2]; o[3] = (bf16)v[3];
            *(v4bf*)&sA[row * LDA_M + col] = o;
        }
    }
    __syncthreads();

    const int wave   = tid >> 5;       // colblock 0..7
    const int lane   = tid & 31;
    const int colloc = lane & 15;
    const int col    = wave * 16 + colloc;
    const int half   = lane >> 4;

    // B fragments held in VGPRs, reused over 4 row-blocks (coalesced 32B/lane loads)
    v16bf bfrag[KS_M];
    const bf16* pb = Pmsg + ((long long)wave * KS_M * 32 + lane) * 16;
#pragma unroll
    for (int ks = 0; ks < KS_M; ++ks)
        bfrag[ks] = *(const v16bf*)(pb + ks * 512);

    const float bias = bmsg[col];
    float* const aggb = agg + (long long)b * N_ * OUT + col;
#pragma unroll
    for (int rb = 0; rb < 4; ++rb) {
        v8f acc = {};
        const bf16* pa = &sA[(rb * 16 + colloc) * LDA_M + half * 8];
#pragma unroll
        for (int ks = 0; ks < KS_M; ++ks) {
            v8bf lo = *(const v8bf*)(pa + ks * 32);
            v8bf hi = *(const v8bf*)(pa + ks * 32 + 16);
            v16bf a;
#pragma unroll
            for (int i = 0; i < 8; ++i) { a[i] = lo[i]; a[i + 8] = hi[i]; }
            acc = __builtin_amdgcn_wmma_f32_16x16x32_bf16(false, a, false, bfrag[ks],
                                                          (short)0, acc, false, false);
        }
        // scatter-add: C layout -> element r is row m = rb*16 + r + half*8, col = wave*16 + (lane&15)
        if (full) {
#pragma unroll
            for (int r = 0; r < 8; ++r) {
                int m = rb * 16 + r + half * 8;
                atomicAdd(aggb + (long long)s_dst[m] * OUT, acc[r] + bias);
            }
        } else {
#pragma unroll
            for (int r = 0; r < 8; ++r) {
                int m = rb * 16 + r + half * 8;
                if (ebase + m < E_)
                    atomicAdd(aggb + (long long)s_dst[m] * OUT, acc[r] + bias);
            }
        }
    }
}

// In-place update: 16 rows of agg (= d_out) per workgroup; out = (agg/cnt) @ W_upd + b_upd
__global__ void __launch_bounds__(256) upd_kernel(
        float* __restrict__ agg, const float* __restrict__ counts,
        const bf16* __restrict__ Pupd, const float* __restrict__ bupd,
        int N_, long long ROWS) {
    __shared__ bf16  sA[16 * LDA_U];
    __shared__ float s_inv[16];

    const int tid = threadIdx.x;
    const long long rbase = (long long)blockIdx.x * 16;
    const bool full = (rbase + 16) <= ROWS;

    if (tid < 16) {
        long long rg = rbase + tid;
        float c = 1.0f;
        if (rg < ROWS) c = counts[(int)(rg % N_)];
        s_inv[tid] = 1.0f / fmaxf(c, 1.0f);
    }
    __syncthreads();

#pragma unroll
    for (int it = 0; it < 2; ++it) {
        int q   = it * 256 + tid;   // 512 float4 chunks
        int row = q >> 5;           // 32 float4 per row
        int col = (q & 31) * 4;
        long long rg = rbase + row;
        v4f v = (v4f)(0.0f);
        if (full || rg < ROWS) v = *(const v4f*)(agg + rg * OUT + col);
        float s = s_inv[row];
        v4bf o; o[0] = (bf16)(v[0] * s); o[1] = (bf16)(v[1] * s);
                o[2] = (bf16)(v[2] * s); o[3] = (bf16)(v[3] * s);
        *(v4bf*)&sA[row * LDA_U + col] = o;
    }
    __syncthreads();

    const int wave   = tid >> 5;
    const int lane   = tid & 31;
    const int colloc = lane & 15;
    const int col    = wave * 16 + colloc;
    const int half   = lane >> 4;

    v8f acc = {};
    const bf16* pa = &sA[colloc * LDA_U + half * 8];
    const bf16* pb = Pupd + ((long long)wave * KS_U * 32 + lane) * 16;
#pragma unroll
    for (int ks = 0; ks < KS_U; ++ks) {
        v8bf lo = *(const v8bf*)(pa + ks * 32);
        v8bf hi = *(const v8bf*)(pa + ks * 32 + 16);
        v16bf a;
#pragma unroll
        for (int i = 0; i < 8; ++i) { a[i] = lo[i]; a[i + 8] = hi[i]; }
        v16bf bf = *(const v16bf*)(pb + ks * 512);
        acc = __builtin_amdgcn_wmma_f32_16x16x32_bf16(false, a, false, bf,
                                                      (short)0, acc, false, false);
    }
    const float bias = bupd[col];
    if (full) {
#pragma unroll
        for (int r = 0; r < 8; ++r)
            agg[(rbase + r + half * 8) * OUT + col] = acc[r] + bias;
    } else {
#pragma unroll
        for (int r = 0; r < 8; ++r) {
            long long rg = rbase + r + half * 8;
            if (rg < ROWS) agg[rg * OUT + col] = acc[r] + bias;
        }
    }
}

extern "C" void kernel_launch(void* const* d_in, const int* in_sizes, int n_in,
                              void* d_out, int out_size, void* d_ws, size_t ws_size,
                              hipStream_t stream) {
    const float* x    = (const float*)d_in[0];
    const int*   eidx = (const int*)d_in[1];
    const float* eatt = (const float*)d_in[2];
    const float* Wmsg = (const float*)d_in[3];
    const float* bmsg = (const float*)d_in[4];
    const float* Wupd = (const float*)d_in[5];
    const float* bupd = (const float*)d_in[6];

    const int B_ = 2;
    const int E_ = in_sizes[1] / 2;
    const int N_ = in_sizes[0] / (B_ * DIM);
    const long long ROWS = (long long)B_ * N_;

    float* agg    = (float*)d_out;   // accumulator, then final output (in-place update)
    float* counts = (float*)d_ws;    // N floats
    size_t off = ((size_t)N_ * sizeof(float) + 255) & ~(size_t)255;
    bf16* Pmsg = (bf16*)((char*)d_ws + off);
    bf16* Pupd = Pmsg + PM_ELEMS;

    const int* src = eidx;
    const int* dst = eidx + E_;

    long long ztot = ROWS * OUT;
    zero_kernel<<<(int)((ztot + 255) / 256), 256, 0, stream>>>(agg, ztot);
    zero_kernel<<<(N_ + 255) / 256, 256, 0, stream>>>(counts, N_);
    pack_weights<<<(PM_ELEMS + PU_ELEMS + 255) / 256, 256, 0, stream>>>(Wmsg, Wupd, Pmsg, Pupd);
    count_kernel<<<(E_ + 255) / 256, 256, 0, stream>>>(dst, counts, E_);

    dim3 g((E_ + TILE_E - 1) / TILE_E, B_);
    msg_kernel<<<g, 256, 0, stream>>>(x, src, dst, eatt, Pmsg, bmsg, agg, N_, E_);
    upd_kernel<<<(int)((ROWS + 15) / 16), 256, 0, stream>>>(agg, counts, Pupd, bupd, N_, ROWS);
}
